// VisionTransformer_4982162063858
// MI455X (gfx1250) — compile-verified, matrix operand order checked by reading, not run on hardware
//
#include <hip/hip_runtime.h>
#include <hip/hip_bf16.h>

// ---------------------------------------------------------------------------
// Types for CDNA5 WMMA (wave32): v_wmma_f32_16x16x32_f16
// ---------------------------------------------------------------------------
typedef __attribute__((ext_vector_type(16))) _Float16 v16h;
typedef __attribute__((ext_vector_type(8)))  _Float16 v8h;
typedef __attribute__((ext_vector_type(8)))  float    v8f;

// Model constants
#define BB   32
#define DD   768
#define LL   12
#define HH   12
#define NN   196
#define DHD  64
#define FFD  3072
#define MM   (BB * NN)        // 6272
#define BHN  (BB * HH)        // 384
#define KA   224              // attention K padded to multiple of 32

// GEMM epilogue flags
#define GF_BIAS  1
#define GF_RESID 2
#define GF_GELU  4
#define GF_F32   8
#define GF_F16   16

// LDS tile: 128 rows x 32 cols, padded to 40 elems/row (80B) so the 16-row
// fragment reads hit 16 distinct banks (stride 20 banks mod 64).
#define LDS_LDA   40
#define TILE_ELEM (128 * LDS_LDA)          // one A or W tile
#define BUF_ELEM  (2 * TILE_ELEM)          // A tile + W tile per buffer

__device__ __forceinline__ v8f wmma16x16x32(v16h a, v16h b, v8f c) {
  return __builtin_amdgcn_wmma_f32_16x16x32_f16(
      /*neg_a=*/false, a, /*neg_b=*/false, b,
      /*c_mod=*/(short)0, c, /*reuse_a=*/false, /*reuse_b=*/false);
}

// Async global->LDS DMA (gfx1250, ASYNCcnt-tracked, ISA 08 section 4):
// each active lane copies 16B from its global address to its LDS address,
// no VGPR round-trip. LDS offset = low 32 bits of the generic pointer.
__device__ __forceinline__ void async_cp16(unsigned lds_off,
                                           const _Float16* g) {
  asm volatile("global_load_async_to_lds_b128 %0, %1, off"
               :: "v"(lds_off), "v"(g) : "memory");
}
__device__ __forceinline__ void wait_async0() {
  asm volatile("s_wait_asynccnt 0" ::: "memory");
}

// Fragment read from an LDS tile row (16-bit A/B layout, ISA 7.12.2):
// lane&15 selects row/col, lane>>4 selects K half; element e maps to
// K = (e>=8?16:0) + 8*half + (e&7)  ->  two contiguous 16B reads.
__device__ __forceinline__ v16h lds_frag(const _Float16* base, int row, int hf) {
  const _Float16* p = base + row * LDS_LDA + (hf << 3);
  v8h lo = *(const v8h*)(p);
  v8h hi = *(const v8h*)(p + 16);
  return __builtin_shufflevector(lo, hi, 0, 1, 2, 3, 4, 5, 6, 7,
                                 8, 9, 10, 11, 12, 13, 14, 15);
}

// ---------------------------------------------------------------------------
// Batched GEMM: C[M,N] = A[M,K] * W[N,K]^T  (+bias)(+resid)(+gelu)
// A, W row-major f16, leading dimension K (multiple of 32).
// Block = 256 threads = 8 waves; block tile 128x128; wave tile 32(M)x64(N)
// = 2x4 WMMA tiles = 8 v_wmma per K-step per wave.
// Tiles are staged via double-buffered async global->LDS DMA: copies for
// tile k+32 are issued right after the barrier and run in the background
// while the WMMAs consume tile k. OOB rows/cols are CLAMPED (not zeroed):
// garbage only reaches output elements the guarded epilogue never stores,
// keeping the K-loop branch-free (EXEC stays all-ones for WMMA).
// ---------------------------------------------------------------------------
__global__ __launch_bounds__(256) void gemm_wmma_kernel(
    const _Float16* __restrict__ A, const _Float16* __restrict__ W,
    const float* __restrict__ bias, const float* __restrict__ resid,
    float* __restrict__ C32, _Float16* __restrict__ C16,
    int M, int N, int K,
    long long strideA, long long strideW, long long strideC, int flags) {
  // [buffer][ A tile | W tile ]  = 2 * 20KB of the 320KB WGP LDS
  __shared__ __align__(16) _Float16 smem[2][BUF_ELEM];

  const long long batch = blockIdx.z;
  A += batch * strideA;
  W += batch * strideW;
  const long long cOff = batch * strideC;

  const int tid  = threadIdx.x;
  const int wave = tid >> 5;
  const int lane = tid & 31;
  const int wm = wave >> 1;            // 0..3  (M direction)
  const int wn = wave & 1;             // 0..1  (N direction)
  const int mWave = wm * 32;           // wave M offset inside block tile
  const int nWave = wn * 64;           // wave N offset inside block tile
  const int r  = lane & 15;
  const int hf = lane >> 4;

  // Cooperative tile-copy mapping: 512 16B chunks per tile, 2 per thread.
  // chunk -> (row = chunk/4, col = (chunk%4)*8); consecutive threads copy
  // consecutive 16B spans of a row (coalesced).
  const int row0 = (tid + 0)   >> 2, col0 = ((tid + 0)   & 3) << 3;
  const int row1 = (tid + 256) >> 2, col1 = ((tid + 256) & 3) << 3;
  const int aRow0 = blockIdx.y * 128 + row0, aRow1 = blockIdx.y * 128 + row1;
  const int wRow0 = blockIdx.x * 128 + row0, wRow1 = blockIdx.x * 128 + row1;
  const _Float16* gA0 = A + (long long)(aRow0 < M ? aRow0 : M - 1) * K + col0;
  const _Float16* gA1 = A + (long long)(aRow1 < M ? aRow1 : M - 1) * K + col1;
  const _Float16* gW0 = W + (long long)(wRow0 < N ? wRow0 : N - 1) * K + col0;
  const _Float16* gW1 = W + (long long)(wRow1 < N ? wRow1 : N - 1) * K + col1;

  const unsigned bufStride = (unsigned)(BUF_ELEM * sizeof(_Float16));
  const unsigned oA0 = (unsigned)(size_t)(&smem[0][0] + row0 * LDS_LDA + col0);
  const unsigned oA1 = (unsigned)(size_t)(&smem[0][0] + row1 * LDS_LDA + col1);
  const unsigned oW0 = oA0 + (unsigned)(TILE_ELEM * sizeof(_Float16));
  const unsigned oW1 = oA1 + (unsigned)(TILE_ELEM * sizeof(_Float16));

  auto issue_tile = [&](int k, int buf) {
    const unsigned bo = (unsigned)buf * bufStride;
    async_cp16(oA0 + bo, gA0 + k);
    async_cp16(oA1 + bo, gA1 + k);
    async_cp16(oW0 + bo, gW0 + k);
    async_cp16(oW1 + bo, gW1 + k);
  };

  v8f acc[2][4] = {};

  issue_tile(0, 0);
  int cur = 0;
  for (int k0 = 0; k0 < K; k0 += 32, cur ^= 1) {
    wait_async0();        // this wave's outstanding copies (into buf[cur]) done
    __syncthreads();      // every wave's copies done; prior reads of buf[cur^1]
                          // finished before their waves reached this barrier
    if (k0 + 32 < K) issue_tile(k0 + 32, cur ^ 1);   // DMA under the WMMAs

    const _Float16* aBase = smem[cur];
    const _Float16* wBase = smem[cur] + TILE_ELEM;
    v16h a0 = lds_frag(aBase, mWave + r, hf);
    v16h a1 = lds_frag(aBase, mWave + 16 + r, hf);
    v16h b0 = lds_frag(wBase, nWave + r, hf);
    v16h b1 = lds_frag(wBase, nWave + 16 + r, hf);
    v16h b2 = lds_frag(wBase, nWave + 32 + r, hf);
    v16h b3 = lds_frag(wBase, nWave + 48 + r, hf);

    acc[0][0] = wmma16x16x32(a0, b0, acc[0][0]);
    acc[0][1] = wmma16x16x32(a0, b1, acc[0][1]);
    acc[0][2] = wmma16x16x32(a0, b2, acc[0][2]);
    acc[0][3] = wmma16x16x32(a0, b3, acc[0][3]);
    acc[1][0] = wmma16x16x32(a1, b0, acc[1][0]);
    acc[1][1] = wmma16x16x32(a1, b1, acc[1][1]);
    acc[1][2] = wmma16x16x32(a1, b2, acc[1][2]);
    acc[1][3] = wmma16x16x32(a1, b3, acc[1][3]);
  }

  // D layout (ISA 7.12.2): element v of lane l is (m = v + 8*(l>>4), n = l&15).
#pragma unroll
  for (int tm = 0; tm < 2; ++tm) {
#pragma unroll
    for (int tn = 0; tn < 4; ++tn) {
      const int n = blockIdx.x * 128 + nWave + tn * 16 + r;
      if (n >= N) continue;
      const float bv = (flags & GF_BIAS) ? bias[n] : 0.0f;
      const v8f a = acc[tm][tn];
#pragma unroll
      for (int v = 0; v < 8; ++v) {
        const int m = blockIdx.y * 128 + mWave + tm * 16 + hf * 8 + v;
        if (m >= M) continue;
        float val = a[v] + bv;
        if (flags & GF_RESID) val += resid[(long long)m * N + n];
        if (flags & GF_GELU)  val = 0.5f * val * (1.0f + erff(val * 0.70710678118f));
        const long long idx = cOff + (long long)m * N + n;
        if (flags & GF_F32) C32[idx] = val;
        if (flags & GF_F16) C16[idx] = (_Float16)val;
      }
    }
  }
}

// ---------------------------------------------------------------------------
// LayerNorm: one block (256 threads) per row of 768.
// ---------------------------------------------------------------------------
__global__ void ln_kernel(const float* __restrict__ x, const float* __restrict__ w,
                          const float* __restrict__ b, _Float16* __restrict__ y16,
                          float* __restrict__ y32) {
  __shared__ float red[256];
  const int row = blockIdx.x;
  const float* xr = x + (long long)row * DD;
  const int t = threadIdx.x;

  float s = 0.0f;
  for (int c = t; c < DD; c += 256) s += xr[c];
  red[t] = s; __syncthreads();
  for (int st = 128; st > 0; st >>= 1) {
    if (t < st) red[t] += red[t + st];
    __syncthreads();
  }
  const float mean = red[0] * (1.0f / DD);
  __syncthreads();

  float vs = 0.0f;
  for (int c = t; c < DD; c += 256) { float d = xr[c] - mean; vs += d * d; }
  red[t] = vs; __syncthreads();
  for (int st = 128; st > 0; st >>= 1) {
    if (t < st) red[t] += red[t + st];
    __syncthreads();
  }
  const float inv = rsqrtf(red[0] * (1.0f / DD) + 1e-5f);

  for (int c = t; c < DD; c += 256) {
    const float v = (xr[c] - mean) * inv * w[c] + b[c];
    if (y16) y16[(long long)row * DD + c] = (_Float16)v;
    if (y32) y32[(long long)row * DD + c] = v;
  }
}

// ---------------------------------------------------------------------------
// Softmax over 196 keys, scale 1/sqrt(64)=0.125, writes f16 padded to 224.
// One block (256 threads) per (b,h,i) score row.
// ---------------------------------------------------------------------------
__global__ void softmax_kernel(const float* __restrict__ S, _Float16* __restrict__ P) {
  __shared__ float red[256];
  const long long bi = blockIdx.x;           // bh*196 + i
  const float* srow = S + bi * NN;
  _Float16* prow = P + bi * KA;
  const int j = threadIdx.x;

  const float v = (j < NN) ? srow[j] * 0.125f : -1e30f;
  red[j] = v; __syncthreads();
  for (int st = 128; st > 0; st >>= 1) {
    if (j < st) red[j] = fmaxf(red[j], red[j + st]);
    __syncthreads();
  }
  const float mx = red[0];
  __syncthreads();

  const float e = (j < NN) ? expf(v - mx) : 0.0f;
  red[j] = e; __syncthreads();
  for (int st = 128; st > 0; st >>= 1) {
    if (j < st) red[j] += red[j + st];
    __syncthreads();
  }
  const float inv = 1.0f / red[0];
  if (j < KA) prow[j] = (_Float16)(e * inv);
}

// ---------------------------------------------------------------------------
// Elementwise helpers
// ---------------------------------------------------------------------------
__global__ void convert_f16_kernel(const float* __restrict__ src,
                                   _Float16* __restrict__ dst, long long n) {
  const long long i = (long long)blockIdx.x * 256 + threadIdx.x;
  if (i < n) dst[i] = (_Float16)src[i];
}

__global__ void fill_zero16_kernel(_Float16* __restrict__ dst, long long n) {
  const long long i = (long long)blockIdx.x * 256 + threadIdx.x;
  if (i < n) dst[i] = (_Float16)0.0f;
}

// im2col for 16x16 patches: A16[(b*196+n), ch*256+ph*16+pw] = x[b,ch,...]
__global__ void im2col_kernel(const float* __restrict__ x, _Float16* __restrict__ A16) {
  const long long i = (long long)blockIdx.x * 256 + threadIdx.x;
  if (i >= (long long)MM * DD) return;
  const int c = (int)(i % DD);
  const int row = (int)(i / DD);
  const int b = row / NN, n = row % NN;
  const int ch = c >> 8, rem = c & 255;
  const int ph = rem >> 4, pw = rem & 15;
  const int pr = n / 14, pc = n % 14;
  const long long src = (((long long)(b * 3 + ch) * 224) + (pr * 16 + ph)) * 224
                      + (pc * 16 + pw);
  A16[i] = (_Float16)x[src];
}

__global__ void add_pos_kernel(float* __restrict__ h, const float* __restrict__ pos) {
  const long long i = (long long)blockIdx.x * 256 + threadIdx.x;
  if (i >= (long long)MM * DD) return;
  const int c = (int)(i % DD);
  const int row = (int)(i / DD);
  h[i] += pos[(long long)(row % NN) * DD + c];
}

// Split qkv (6272 x 2304 f32) into Qh/Kh (bh,n,dh) f16 and Vt (bh,dh,n-pad224) f16
__global__ void extract_qkv_kernel(const float* __restrict__ qkv,
                                   _Float16* __restrict__ Qh,
                                   _Float16* __restrict__ Kh,
                                   _Float16* __restrict__ Vt) {
  const long long i = (long long)blockIdx.x * 256 + threadIdx.x;
  if (i >= (long long)BHN * NN * DHD) return;
  const int dh = (int)(i % DHD);
  const long long t = i / DHD;
  const int n = (int)(t % NN);
  const int bh = (int)(t / NN);
  const int b = bh / HH, hh = bh % HH;
  const long long rbase = ((long long)(b * NN + n)) * (3 * DD) + hh * DHD + dh;
  Qh[i] = (_Float16)qkv[rbase];
  Kh[i] = (_Float16)qkv[rbase + DD];
  Vt[((long long)bh * DHD + dh) * KA + n] = (_Float16)qkv[rbase + 2 * DD];
}

// Merge heads: o16[(b*196+n), h*64+dh] = O32[(b*H+h), n, dh]
__global__ void merge_o_kernel(const float* __restrict__ O32, _Float16* __restrict__ o16) {
  const long long i = (long long)blockIdx.x * 256 + threadIdx.x;
  if (i >= (long long)MM * DD) return;
  const int c = (int)(i % DD);
  const int row = (int)(i / DD);
  const int b = row / NN, n = row % NN;
  const int hh = c >> 6, dh = c & 63;
  o16[i] = (_Float16)O32[(((long long)(b * HH + hh)) * NN + n) * DHD + dh];
}

// ---------------------------------------------------------------------------
// Host driver
// ---------------------------------------------------------------------------
extern "C" void kernel_launch(void* const* d_in, const int* in_sizes, int n_in,
                              void* d_out, int out_size, void* d_ws, size_t ws_size,
                              hipStream_t stream) {
  const float* x       = (const float*)d_in[0];
  const float* patch_w = (const float*)d_in[1];
  const float* patch_b = (const float*)d_in[2];
  const float* pos     = (const float*)d_in[3];
  const float* ln1_w   = (const float*)d_in[4];
  const float* ln1_b   = (const float*)d_in[5];
  const float* qkv_w   = (const float*)d_in[6];
  const float* qkv_b   = (const float*)d_in[7];
  const float* proj_w  = (const float*)d_in[8];
  const float* proj_b  = (const float*)d_in[9];
  const float* ln2_w   = (const float*)d_in[10];
  const float* ln2_b   = (const float*)d_in[11];
  const float* fc1_w   = (const float*)d_in[12];
  const float* fc1_b   = (const float*)d_in[13];
  const float* fc2_w   = (const float*)d_in[14];
  const float* fc2_b   = (const float*)d_in[15];
  const float* lnf_w   = (const float*)d_in[16];
  const float* lnf_b   = (const float*)d_in[17];
  (void)in_sizes; (void)n_in; (void)out_size; (void)ws_size;

  // Workspace layout (256B aligned regions, ~282 MB total)
  char* ws = (char*)d_ws;
  size_t off = 0;
  auto carve = [&](size_t bytes) -> void* {
    void* p = ws + off;
    off += (bytes + 255) & ~(size_t)255;
    return p;
  };
  float*    h32   = (float*)   carve((size_t)MM * DD * 4);
  _Float16* y16   = (_Float16*)carve((size_t)MM * DD * 2);
  float*    qkv32 = (float*)   carve((size_t)MM * 3 * DD * 4);
  _Float16* Qh    = (_Float16*)carve((size_t)BHN * NN * DHD * 2);
  _Float16* Kh    = (_Float16*)carve((size_t)BHN * NN * DHD * 2);
  _Float16* Vt    = (_Float16*)carve((size_t)BHN * DHD * KA * 2);
  float*    S32   = (float*)   carve((size_t)BHN * NN * NN * 4);
  _Float16* P16   = (_Float16*)carve((size_t)BHN * NN * KA * 2);
  float*    O32   = (float*)   carve((size_t)BHN * NN * DHD * 4);
  _Float16* o16   = (_Float16*)carve((size_t)MM * DD * 2);
  _Float16* ff16  = (_Float16*)carve((size_t)MM * FFD * 2);
  _Float16* W16   = (_Float16*)carve((size_t)FFD * DD * 2);

  auto gemm = [&](const _Float16* A, const _Float16* W, const float* bias,
                  const float* resid, float* C32, _Float16* C16,
                  int M, int N, int K, long long sA, long long sW, long long sC,
                  int batch, int flags) {
    dim3 grid((N + 127) / 128, (M + 127) / 128, batch);
    gemm_wmma_kernel<<<grid, 256, 0, stream>>>(A, W, bias, resid, C32, C16,
                                               M, N, K, sA, sW, sC, flags);
  };
  auto blocks1d = [](long long n) { return (unsigned)((n + 255) / 256); };

  const long long elemMD  = (long long)MM * DD;
  const long long elemQKV = (long long)BHN * NN * DHD;

  // --- patch embed: im2col + GEMM + pos add ---
  im2col_kernel<<<blocks1d(elemMD), 256, 0, stream>>>(x, y16);
  convert_f16_kernel<<<blocks1d((long long)DD * DD), 256, 0, stream>>>(
      patch_w, W16, (long long)DD * DD);
  gemm(y16, W16, patch_b, nullptr, h32, nullptr, MM, DD, DD, 0, 0, 0, 1,
       GF_BIAS | GF_F32);
  add_pos_kernel<<<blocks1d(elemMD), 256, 0, stream>>>(h32, pos);

  for (int i = 0; i < LL; ++i) {
    // --- attention ---
    ln_kernel<<<MM, 256, 0, stream>>>(h32, ln1_w + (long long)i * DD,
                                      ln1_b + (long long)i * DD, y16, nullptr);
    convert_f16_kernel<<<blocks1d((long long)3 * DD * DD), 256, 0, stream>>>(
        qkv_w + (long long)i * 3 * DD * DD, W16, (long long)3 * DD * DD);
    gemm(y16, W16, qkv_b + (long long)i * 3 * DD, nullptr, qkv32, nullptr,
         MM, 3 * DD, DD, 0, 0, 0, 1, GF_BIAS | GF_F32);

    fill_zero16_kernel<<<blocks1d((long long)BHN * DHD * KA), 256, 0, stream>>>(
        Vt, (long long)BHN * DHD * KA);
    extract_qkv_kernel<<<blocks1d(elemQKV), 256, 0, stream>>>(qkv32, Qh, Kh, Vt);

    // scores = Q @ K^T (batched over 384 heads), softmax, O = P @ V
    gemm(Qh, Kh, nullptr, nullptr, S32, nullptr, NN, NN, DHD,
         (long long)NN * DHD, (long long)NN * DHD, (long long)NN * NN, BHN, GF_F32);
    softmax_kernel<<<BHN * NN, 256, 0, stream>>>(S32, P16);
    gemm(P16, Vt, nullptr, nullptr, O32, nullptr, NN, DHD, KA,
         (long long)NN * KA, (long long)DHD * KA, (long long)NN * DHD, BHN, GF_F32);
    merge_o_kernel<<<blocks1d(elemMD), 256, 0, stream>>>(O32, o16);

    convert_f16_kernel<<<blocks1d((long long)DD * DD), 256, 0, stream>>>(
        proj_w + (long long)i * DD * DD, W16, (long long)DD * DD);
    gemm(o16, W16, proj_b + (long long)i * DD, h32, h32, nullptr,
         MM, DD, DD, 0, 0, 0, 1, GF_BIAS | GF_RESID | GF_F32);

    // --- MLP ---
    ln_kernel<<<MM, 256, 0, stream>>>(h32, ln2_w + (long long)i * DD,
                                      ln2_b + (long long)i * DD, y16, nullptr);
    convert_f16_kernel<<<blocks1d((long long)FFD * DD), 256, 0, stream>>>(
        fc1_w + (long long)i * FFD * DD, W16, (long long)FFD * DD);
    gemm(y16, W16, fc1_b + (long long)i * FFD, nullptr, nullptr, ff16,
         MM, FFD, DD, 0, 0, 0, 1, GF_BIAS | GF_GELU | GF_F16);
    convert_f16_kernel<<<blocks1d((long long)DD * FFD), 256, 0, stream>>>(
        fc2_w + (long long)i * DD * FFD, W16, (long long)DD * FFD);
    gemm(ff16, W16, fc2_b + (long long)i * DD, h32, h32, nullptr,
         MM, DD, FFD, 0, 0, 0, 1, GF_BIAS | GF_RESID | GF_F32);
  }

  // --- final LN -> d_out (f32) ---
  ln_kernel<<<MM, 256, 0, stream>>>(h32, lnf_w, lnf_b, nullptr, (float*)d_out);
}
